// Net_5772436046568
// MI455X (gfx1250) — 8-bit matrix operand order repaired and checked by reading, compiled, not run on hardware
//
#include <hip/hip_runtime.h>
#include <hip/hip_bf16.h>
#include <math.h>

typedef int v8i __attribute__((ext_vector_type(8)));
typedef int v4i_ __attribute__((vector_size(16)));

#define BNN_EPS 1e-5f
#define ROWB 80

#if defined(__has_builtin)
#if __has_builtin(__builtin_amdgcn_global_load_async_to_lds_b128)
#define HAVE_ASYNC_LDS 1
#endif
#if __has_builtin(__builtin_amdgcn_s_wait_asynccnt)
#define HAVE_WAIT_ASYNC 1
#endif
#endif

__device__ __forceinline__ signed char fsign(float x) {
    return (signed char)((x > 0.f) ? 1 : ((x < 0.f) ? -1 : 0));
}
__device__ __forceinline__ signed char isign(int x) {
    return (signed char)((x > 0) ? 1 : ((x < 0) ? -1 : 0));
}
__device__ __forceinline__ signed char i8max(signed char a, signed char b) {
    return a > b ? a : b;
}

__device__ __forceinline__ void copy16_async(const signed char* g, void* l) {
#if HAVE_ASYNC_LDS
    __builtin_amdgcn_global_load_async_to_lds_b128(
        (__attribute__((address_space(1))) v4i_*)g,
        (__attribute__((address_space(3))) v4i_*)l, 0, 0);
#else
    *(int4*)l = *(const int4*)g;
#endif
}
__device__ __forceinline__ void async_wait() {
#if HAVE_WAIT_ASYNC
    __builtin_amdgcn_s_wait_asynccnt(0);
#elif HAVE_ASYNC_LDS
    asm volatile("s_wait_asynccnt 0x0" ::: "memory");
#endif
}

__global__ void binw_conv(const float* __restrict__ w, signed char* __restrict__ wb,
                          int Cout, int Cin) {
    int i = blockIdx.x * blockDim.x + threadIdx.x;
    int total = Cout * Cin * 9;
    if (i >= total) return;
    int kx = i % 3;
    int t  = i / 3;
    int ky = t % 3;
    int t2 = t / 3;
    int ci = t2 % Cin;
    int co = t2 / Cin;
    wb[(size_t)co * (9 * Cin) + (ky * 3 + kx) * Cin + ci] = fsign(w[i]);
}

__global__ void binw_fc1(const float* __restrict__ w, signed char* __restrict__ wb) {
    int i = blockIdx.x * blockDim.x + threadIdx.x;
    if (i >= 1024 * 8192) return;
    int k = i % 8192;
    int n = i / 8192;
    int c  = k >> 4;
    int yx = k & 15;
    wb[(size_t)n * 8192 + yx * 512 + c] = fsign(w[i]);
}

__global__ void binw_fc2(const float* __restrict__ w, signed char* __restrict__ wb) {
    int i = blockIdx.x * blockDim.x + threadIdx.x;
    if (i >= 10 * 1024) return;
    wb[i] = fsign(w[i]);
}

__global__ __launch_bounds__(128)
void conv0_bn_sign(const float* __restrict__ x, const float* __restrict__ w,
                   const float* __restrict__ g, const float* __restrict__ bb_,
                   const float* __restrict__ mm, const float* __restrict__ vv,
                   signed char* __restrict__ out) {
    const int H = 32, W = 32;
    int pix = blockIdx.x;
    int co  = threadIdx.x;
    int xw = pix & 31;
    int yh = (pix >> 5) & 31;
    int bn = pix >> 10;

    __shared__ float patch[3][3][3];
    if (threadIdx.x < 27) {
        int ci = threadIdx.x / 9;
        int r  = threadIdx.x % 9;
        int ky = r / 3, kx = r % 3;
        int yy = yh + ky - 1, xx = xw + kx - 1;
        float val = 0.f;
        if (yy >= 0 && yy < H && xx >= 0 && xx < W)
            val = x[(((size_t)bn * 3 + ci) * H + yy) * W + xx];
        patch[ci][ky][kx] = val;
    }
    __syncthreads();

    float acc = 0.f;
#pragma unroll
    for (int ci = 0; ci < 3; ci++)
#pragma unroll
        for (int ky = 0; ky < 3; ky++)
#pragma unroll
            for (int kx = 0; kx < 3; kx++)
                acc = fmaf(patch[ci][ky][kx], w[((co * 3 + ci) * 3 + ky) * 3 + kx], acc);

    float inv = g[co] * rsqrtf(vv[co] + BNN_EPS);
    float val = acc * inv + (bb_[co] - mm[co] * inv);
    out[(size_t)pix * 128 + co] = fsign(val);
}

__device__ __forceinline__ void wmma_step8(const signed char* __restrict__ As,
                                           const signed char* __restrict__ Bs,
                                           int lane, int waveM, int waveN,
                                           v8i acc[2][4]) {
    const int half = (lane >= 16) ? 1 : 0;
    const int rA   = (lane & 15);
    const int kb   = half ? 16 : 0;
#pragma unroll
    for (int sm = 0; sm < 2; sm++) {
        const signed char* ap = As + (waveM + sm * 16 + rA) * ROWB + kb;
        int4 a03 = *(const int4*)(ap + 0);
        int4 a47 = *(const int4*)(ap + 32);
        v8i a = {a03.x, a03.y, a03.z, a03.w, a47.x, a47.y, a47.z, a47.w};
#pragma unroll
        for (int sn = 0; sn < 4; sn++) {
            const signed char* bp = Bs + (waveN + sn * 16 + rA) * ROWB + half * 16;
            int4 b03 = *(const int4*)(bp + 0);
            int4 b47 = *(const int4*)(bp + 32);
            v8i bf = {b03.x, b03.y, b03.z, b03.w, b47.x, b47.y, b47.z, b47.w};
            acc[sm][sn] = __builtin_amdgcn_wmma_i32_16x16x64_iu8(
                true, a, true, bf, acc[sm][sn], false, false);
        }
    }
}

__global__ __launch_bounds__(256)
void bconv_wmma(const signed char* __restrict__ act, const signed char* __restrict__ wb,
                signed char* __restrict__ out, int wsh, int hsh, int Cin, int Cout) {
    __shared__ int As4[2][128 * (ROWB / 4)];
    __shared__ int Bs4[2][128 * (ROWB / 4)];

    const int W = 1 << wsh, H = 1 << hsh;
    const int tid  = threadIdx.x;
    const int lane = tid & 31;
    const int wid  = tid >> 5;
    const int Mbase = blockIdx.x * 128;
    const int Nbase = blockIdx.y * 128;
    const int Ktot = 9 * Cin;
    const int waveM = (wid >> 1) * 32;
    const int waveN = (wid & 1) * 64;

    const int ch = (tid & 3) * 16;
    const int r0 = tid >> 2;
    int ax[2], ay[2], ab[2];
#pragma unroll
    for (int h = 0; h < 2; h++) {
        int m = Mbase + r0 + h * 64;
        ax[h] = m & (W - 1);
        ay[h] = (m >> wsh) & (H - 1);
        ab[h] = m >> (wsh + hsh);
    }

    const int nchunks = Cin >> 6;
    const int total   = 9 * nchunks;

    auto stage = [&](int t, int buf) {
        const int o  = t / nchunks;
        const int c0 = (t - o * nchunks) << 6;
        const int dy = o / 3 - 1;
        const int dx = o % 3 - 1;
#pragma unroll
        for (int h = 0; h < 2; h++) {
            int yy = ay[h] + dy, xx = ax[h] + dx;
            void* dst = (char*)As4[buf] + (r0 + h * 64) * ROWB + ch;
            if ((unsigned)yy < (unsigned)H && (unsigned)xx < (unsigned)W)
                copy16_async(act + ((size_t)(((ab[h] << hsh) + yy) << wsh) + xx) * Cin + c0 + ch,
                             dst);
            else
                *(int4*)dst = make_int4(0, 0, 0, 0);
        }
#pragma unroll
        for (int h = 0; h < 2; h++) {
            int n = Nbase + r0 + h * 64;
            copy16_async(wb + (size_t)n * Ktot + o * Cin + c0 + ch,
                         (char*)Bs4[buf] + (r0 + h * 64) * ROWB + ch);
        }
    };

    v8i zero = {0, 0, 0, 0, 0, 0, 0, 0};
    v8i acc[2][4];
#pragma unroll
    for (int i = 0; i < 2; i++)
#pragma unroll
        for (int j = 0; j < 4; j++) acc[i][j] = zero;

    stage(0, 0);
    for (int t = 0; t < total; t++) {
        const int cur = t & 1;
        async_wait();
        __syncthreads();
        if (t + 1 < total) stage(t + 1, cur ^ 1);
        wmma_step8((const signed char*)As4[cur], (const signed char*)Bs4[cur],
                   lane, waveM, waveN, acc);
    }

#pragma unroll
    for (int sm = 0; sm < 2; sm++)
#pragma unroll
        for (int sn = 0; sn < 4; sn++)
#pragma unroll
            for (int r = 0; r < 8; r++) {
                int mrow = Mbase + waveM + sm * 16 + ((lane < 16) ? r : 8 + r);
                int ncol = Nbase + waveN + sn * 16 + (lane & 15);
                out[(size_t)mrow * Cout + ncol] = isign(acc[sm][sn][r]);
            }
}

__global__ __launch_bounds__(256)
void bfc_wmma(const signed char* __restrict__ A, const signed char* __restrict__ Bt,
              int* __restrict__ out, int M, int N, int K) {
    __shared__ int As4[2][128 * (ROWB / 4)];
    __shared__ int Bs4[2][128 * (ROWB / 4)];

    const int tid  = threadIdx.x;
    const int lane = tid & 31;
    const int wid  = tid >> 5;
    const int Mbase = blockIdx.x * 128;
    const int Nbase = blockIdx.y * 128;
    const int waveM = (wid >> 1) * 32;
    const int waveN = (wid & 1) * 64;

    const int ch = (tid & 3) * 16;
    const int r0 = tid >> 2;

    auto stage = [&](int t, int buf) {
        const int k0 = t << 6;
#pragma unroll
        for (int h = 0; h < 2; h++) {
            copy16_async(A + (size_t)(Mbase + r0 + h * 64) * K + k0 + ch,
                         (char*)As4[buf] + (r0 + h * 64) * ROWB + ch);
            copy16_async(Bt + (size_t)(Nbase + r0 + h * 64) * K + k0 + ch,
                         (char*)Bs4[buf] + (r0 + h * 64) * ROWB + ch);
        }
    };

    v8i zero = {0, 0, 0, 0, 0, 0, 0, 0};
    v8i acc[2][4];
#pragma unroll
    for (int i = 0; i < 2; i++)
#pragma unroll
        for (int j = 0; j < 4; j++) acc[i][j] = zero;

    const int total = K >> 6;
    stage(0, 0);
    for (int t = 0; t < total; t++) {
        const int cur = t & 1;
        async_wait();
        __syncthreads();
        if (t + 1 < total) stage(t + 1, cur ^ 1);
        wmma_step8((const signed char*)As4[cur], (const signed char*)Bs4[cur],
                   lane, waveM, waveN, acc);
    }

#pragma unroll
    for (int sm = 0; sm < 2; sm++)
#pragma unroll
        for (int sn = 0; sn < 4; sn++)
#pragma unroll
            for (int r = 0; r < 8; r++) {
                int mrow = Mbase + waveM + sm * 16 + ((lane < 16) ? r : 8 + r);
                int ncol = Nbase + waveN + sn * 16 + (lane & 15);
                out[(size_t)mrow * N + ncol] = acc[sm][sn][r];
            }
}

__global__ void maxpool_i8(const signed char* __restrict__ in, signed char* __restrict__ out,
                           int B, int H, int W, int C) {
    int i = blockIdx.x * blockDim.x + threadIdx.x;
    int Ho = H / 2, Wo = W / 2;
    int total = B * Ho * Wo * C;
    if (i >= total) return;
    int c  = i % C;
    int t  = i / C;
    int xo = t % Wo; t /= Wo;
    int yo = t % Ho;
    int bn = t / Ho;
    const signed char* p  = in + ((size_t)(bn * H + yo * 2) * W + xo * 2) * C + c;
    const signed char* p2 = p + (size_t)W * C;
    out[i] = i8max(i8max(p[0], p[C]), i8max(p2[0], p2[C]));
}

__global__ void bn1d_sign(const int* __restrict__ acc,
                          const float* __restrict__ g, const float* __restrict__ bb_,
                          const float* __restrict__ mm, const float* __restrict__ vv,
                          signed char* __restrict__ out) {
    int i = blockIdx.x * blockDim.x + threadIdx.x;
    if (i >= 256 * 1024) return;
    int n = i & 1023;
    float inv = g[n] * rsqrtf(vv[n] + BNN_EPS);
    float val = (float)acc[i] * inv + (bb_[n] - mm[n] * inv);
    out[i] = fsign(val);
}

__global__ __launch_bounds__(32)
void fc2_lsm(const signed char* __restrict__ a, const signed char* __restrict__ wb,
             float* __restrict__ out) {
    int bn   = blockIdx.x;
    int lane = threadIdx.x;
    __shared__ float logits[10];
    if (lane < 10) {
        int s = 0;
        const signed char* ap = a + (size_t)bn * 1024;
        const signed char* wp = wb + (size_t)lane * 1024;
        for (int k = 0; k < 1024; k++) s += (int)ap[k] * (int)wp[k];
        logits[lane] = (float)s;
    }
    __syncthreads();
    if (lane == 0) {
        float mx = -1e30f;
        for (int n = 0; n < 10; n++) mx = fmaxf(mx, logits[n]);
        float sum = 0.f;
        for (int n = 0; n < 10; n++) sum += expf(logits[n] - mx);
        float lse = logf(sum) + mx;
        for (int n = 0; n < 10; n++) out[(size_t)bn * 10 + n] = logits[n] - lse;
    }
}

extern "C" void kernel_launch(void* const* d_in, const int* in_sizes, int n_in,
                              void* d_out, int out_size, void* d_ws, size_t ws_size,
                              hipStream_t stream) {
    (void)in_sizes; (void)n_in; (void)out_size; (void)ws_size;

    const float* x     = (const float*)d_in[0];
    const float* w0    = (const float*)d_in[1];
    const float* bn1_g = (const float*)d_in[2];
    const float* bn1_b = (const float*)d_in[3];
    const float* bn1_m = (const float*)d_in[4];
    const float* bn1_v = (const float*)d_in[5];
    const float* w1    = (const float*)d_in[6];
    const float* w2    = (const float*)d_in[7];
    const float* w3    = (const float*)d_in[8];
    const float* w4    = (const float*)d_in[9];
    const float* w5    = (const float*)d_in[10];
    const float* fc1_w = (const float*)d_in[11];
    const float* bnf_g = (const float*)d_in[12];
    const float* bnf_b = (const float*)d_in[13];
    const float* bnf_m = (const float*)d_in[14];
    const float* bnf_v = (const float*)d_in[15];
    const float* fc2_w = (const float*)d_in[16];

    const int B = 256;

    signed char* buf0 = (signed char*)d_ws;
    signed char* buf1 = buf0 + 33554432;
    signed char* w1b  = buf1 + 33554432;
    signed char* w2b  = w1b + 147456;
    signed char* w3b  = w2b + 294912;
    signed char* w4b  = w3b + 589824;
    signed char* w5b  = w4b + 1179648;
    signed char* fc1b = w5b + 2359296;
    signed char* fc2b = fc1b + 8388608;
    int*         fcacc = (int*)(fc2b + 10240);

    binw_conv<<<(128 * 128 * 9 + 255) / 256, 256, 0, stream>>>(w1, w1b, 128, 128);
    binw_conv<<<(256 * 128 * 9 + 255) / 256, 256, 0, stream>>>(w2, w2b, 256, 128);
    binw_conv<<<(256 * 256 * 9 + 255) / 256, 256, 0, stream>>>(w3, w3b, 256, 256);
    binw_conv<<<(512 * 256 * 9 + 255) / 256, 256, 0, stream>>>(w4, w4b, 512, 256);
    binw_conv<<<(512 * 512 * 9 + 255) / 256, 256, 0, stream>>>(w5, w5b, 512, 512);
    binw_fc1<<<(1024 * 8192 + 255) / 256, 256, 0, stream>>>(fc1_w, fc1b);
    binw_fc2<<<(10 * 1024 + 255) / 256, 256, 0, stream>>>(fc2_w, fc2b);

    conv0_bn_sign<<<B * 32 * 32, 128, 0, stream>>>(x, w0, bn1_g, bn1_b, bn1_m, bn1_v, buf0);

    bconv_wmma<<<dim3(2048, 1), 256, 0, stream>>>(buf0, w1b, buf1, 5, 5, 128, 128);
    maxpool_i8<<<(B * 16 * 16 * 128 + 255) / 256, 256, 0, stream>>>(buf1, buf0, B, 32, 32, 128);

    bconv_wmma<<<dim3(512, 2), 256, 0, stream>>>(buf0, w2b, buf1, 4, 4, 128, 256);
    bconv_wmma<<<dim3(512, 2), 256, 0, stream>>>(buf1, w3b, buf0, 4, 4, 256, 256);
    maxpool_i8<<<(B * 8 * 8 * 256 + 255) / 256, 256, 0, stream>>>(buf0, buf1, B, 16, 16, 256);

    bconv_wmma<<<dim3(128, 4), 256, 0, stream>>>(buf1, w4b, buf0, 3, 3, 256, 512);
    bconv_wmma<<<dim3(128, 4), 256, 0, stream>>>(buf0, w5b, buf1, 3, 3, 512, 512);
    maxpool_i8<<<(B * 4 * 4 * 512 + 255) / 256, 256, 0, stream>>>(buf1, buf0, B, 8, 8, 512);

    bfc_wmma<<<dim3(2, 8), 256, 0, stream>>>(buf0, fc1b, fcacc, 256, 1024, 8192);
    bn1d_sign<<<(256 * 1024 + 255) / 256, 256, 0, stream>>>(fcacc, bnf_g, bnf_b, bnf_m, bnf_v,
                                                            buf1);

    fc2_lsm<<<B, 32, 0, stream>>>(buf1, fc2b, (float*)d_out);
}
